// InvertISDPvec_62603443307065
// MI455X (gfx1250) — compile-verified, matrix operand order checked by reading, not run on hardware
//
#include <hip/hip_runtime.h>
#include <hip/hip_bf16.h>

// ---------------------------------------------------------------------------
// InvertISDP coupling layer for MI455X (gfx1250, wave32, WMMA + TDM/async).
//
// Math reductions (exact):
//   * attention outputs only used via SA@w  =>  per-head scalar
//     vw[k] = y[k,:] @ (Wv^T w)_head; attn@V GEMM and V projection vanish.
//   * logdet = log|exp(SA1@wA)| * 512 = (SA1@wA) * 512.
//   * ones-column of y folded into effective biases.
// Heavy work left: (50176x640)@(640x2048) fp16 WMMA GEMM + QK^T scores.
// Data movement: TDM tensor_load_to_lds (preferred) or async-to-LDS copies,
// double-buffered so tile DMA overlaps the 8 WMMAs per K-slice.
// ---------------------------------------------------------------------------

typedef __attribute__((ext_vector_type(16))) _Float16 v16h;
typedef __attribute__((ext_vector_type(8)))  _Float16 v8h;
typedef __attribute__((ext_vector_type(8)))  float    v8f;
typedef int v4i __attribute__((vector_size(16)));   // gcc-style, matches builtin pointee

#define N_TOK   50176      // B * HW
#define HW      196
#define BATCH   256
#define CIN     1024
#define KD      640        // DS + COND (ones column folded into bias)
#define QKVD    641
#define COND_   128
#define LDY     40         // LDS row stride (halves) for GEMM tiles, bank-safe
#define SLD     210        // LDS row stride (floats) for score rows, bank-safe
#define NKTILE  20         // 640 / 32

#ifndef __has_builtin
#define __has_builtin(x) 0
#endif

#if __has_builtin(__builtin_amdgcn_tensor_load_to_lds) && \
    __has_builtin(__builtin_amdgcn_s_wait_tensorcnt)
#define HAVE_TDM 1
#else
#define HAVE_TDM 0
#endif

#if __has_builtin(__builtin_amdgcn_global_load_async_to_lds_b128) && \
    __has_builtin(__builtin_amdgcn_s_wait_asynccnt)
#define HAVE_ASYNC 1
#else
#define HAVE_ASYNC 0
#endif

#define AS_GLOBAL __attribute__((address_space(1)))
#define AS_LOCAL  __attribute__((address_space(3)))

#define WMMA_F16(a, b, c) \
  __builtin_amdgcn_wmma_f32_16x16x32_f16(false, (a), false, (b), (short)0, (c), false, false)

// ---- 16B global -> LDS copy: async when available, else register bounce ----
__device__ __forceinline__ void cp_b128(_Float16* ldst, const _Float16* gsrc) {
#if HAVE_ASYNC
  __builtin_amdgcn_global_load_async_to_lds_b128(
      (AS_GLOBAL v4i*)(void*)gsrc, (AS_LOCAL v4i*)(void*)ldst, 0, 0);
#else
  *(v8h*)ldst = *(const v8h*)gsrc;
#endif
}
__device__ __forceinline__ void cp_b128_wait_all() {
#if HAVE_ASYNC
  __builtin_amdgcn_s_wait_asynccnt(0);
#endif
}

#if HAVE_TDM
typedef __attribute__((ext_vector_type(4))) unsigned int u32x4;
typedef __attribute__((ext_vector_type(8))) int          i32x8;
typedef __attribute__((ext_vector_type(4))) int          i32x4;

// One TDM descriptor: 128 rows x 64B tile out of a row-major tensor with
// 1280B row stride, landing in LDS with +16B padding after every 64B row
// (pad_interval=16 DWORDs, pad_amount=4 DWORDs) => 80B LDS row stride (LDY).
// D# layout per CDNA5 ISA sec 8.3/8.4 (group0: count/lds/global/type,
// group1: data_size=1B, dims/strides/tile dims, pad fields).
__device__ __forceinline__ void tdm_load_tile(const _Float16* gsrc, _Float16* ldst) {
  unsigned long long ga = (unsigned long long)(const void*)gsrc;
  unsigned lds_addr = (unsigned)(unsigned long long)(void*)ldst; // flat[31:0] = LDS offset
  u32x4 g0;
  g0[0] = 1u;                                          // count=1, user descriptor
  g0[1] = lds_addr;                                    // lds_addr
  g0[2] = (unsigned)ga;                                // global_addr[31:0]
  g0[3] = (unsigned)((ga >> 32) & 0x01FFFFFFull)       // global_addr[56:32]
          | (2u << 30);                                // type=2 ("image")
  i32x8 g1;
  g1[0] = (1 << 20) | (3 << 22) | (3 << 25);           // pad_en, interval=16DW, amount=4DW
  g1[1] = (int)((1280u & 0xFFFFu) << 16);              // atomic_addr=0 | tensor_dim0 lo16
  g1[2] = (int)((1280u >> 16) | (((unsigned)N_TOK & 0xFFFFu) << 16)); // dim0 hi | dim1 lo
  g1[3] = (int)(((unsigned)N_TOK >> 16) | (64u << 16));               // dim1 hi | tile_dim0=64B
  g1[4] = 128;                                         // tile_dim1=128 rows, tile_dim2=0
  g1[5] = 1280;                                        // tensor_dim0_stride = 1280B (lo32)
  g1[6] = 0;                                           // stride hi | dim1_stride lo (unused, 2D)
  g1[7] = 0;
  i32x4 z = {};                                        // groups 2/3: unused (<=2D tile)
#if __clang_major__ >= 23
  i32x8 z8 = {};
  __builtin_amdgcn_tensor_load_to_lds(g0, g1, z, z, z8, 0);
#else
  __builtin_amdgcn_tensor_load_to_lds(g0, g1, z, z, 0);
#endif
}
#endif  // HAVE_TDM

// Load a 16x32 fp16 WMMA fragment from a row-major [row][k] matrix.
// A-matrix: rows = M.  B-matrix (K x 16 logical): rows = N, source stored [N][K].
// Layout per ISA 7.12.2: lanes 0-15 row=lane, K 0..7 & 16..23;
//                        lanes 16-31 row=lane-16, K 8..15 & 24..31.
__device__ __forceinline__ v16h ld_frag(const _Float16* base, int ld, int row0,
                                        int rowmax, int k0, int lane) {
  int r = row0 + (lane & 15);
  r = r > rowmax ? rowmax : r;
  int kb = (lane & 16) ? 8 : 0;
  const _Float16* p = base + (size_t)r * ld + k0 + kb;
  v8h lo = *(const v8h*)(p);
  v8h hi = *(const v8h*)(p + 16);
  v16h o;
#pragma unroll
  for (int i = 0; i < 8; ++i) { o[i] = lo[i]; o[i + 8] = hi[i]; }
  return o;
}

// ---------------------------------------------------------------------------
// 1) y (fp16): [50176][640] = [inputs[:, :512] | cond]
// ---------------------------------------------------------------------------
__global__ __launch_bounds__(256) void k_build_y(const float* __restrict__ inp,
                                                 const float* __restrict__ cond,
                                                 _Float16* __restrict__ yh) {
  int idx = blockIdx.x * 256 + threadIdx.x;          // < 50176*80, exact grid
  int n  = idx / 80;
  int cc = (idx % 80) * 8;
  const float* src = (cc < 512) ? (inp + (size_t)n * CIN + cc)
                                : (cond + (size_t)n * COND_ + (cc - 512));
  float4 f0 = *(const float4*)(src);
  float4 f1 = *(const float4*)(src + 4);
  v8h o;
  o[0] = (_Float16)f0.x; o[1] = (_Float16)f0.y; o[2] = (_Float16)f0.z; o[3] = (_Float16)f0.w;
  o[4] = (_Float16)f1.x; o[5] = (_Float16)f1.y; o[6] = (_Float16)f1.z; o[7] = (_Float16)f1.w;
  *(v8h*)(yh + (size_t)n * KD + cc) = o;
}

// ---------------------------------------------------------------------------
// 2) Pack [Wq1|Wk1|Wq2|Wk2] -> fp16 [2048][640]
// ---------------------------------------------------------------------------
__global__ __launch_bounds__(256) void k_pack_w(const float* __restrict__ Wq1,
                                                const float* __restrict__ Wk1,
                                                const float* __restrict__ Wq2,
                                                const float* __restrict__ Wk2,
                                                _Float16* __restrict__ wcat) {
  int idx = blockIdx.x * 256 + threadIdx.x;          // < 2048*80
  if (idx >= 2048 * 80) return;
  int row = idx / 80;
  int cc  = (idx % 80) * 8;
  int g = row >> 9, a = row & 511;
  const float* W = (g == 0) ? Wq1 : (g == 1) ? Wk1 : (g == 2) ? Wq2 : Wk2;
  v8h o;
#pragma unroll
  for (int i = 0; i < 8; ++i) o[i] = (_Float16)W[(size_t)a * QKVD + cc + i];
  *(v8h*)(wcat + (size_t)row * KD + cc) = o;
}

// effective biases: beff[a] = b[a] + W[a, 640]  (ones column fold)
__global__ __launch_bounds__(256) void k_beff(const float* __restrict__ Wq1,
                                              const float* __restrict__ Wk1,
                                              const float* __restrict__ Wq2,
                                              const float* __restrict__ Wk2,
                                              const float* __restrict__ bq1,
                                              const float* __restrict__ bk1,
                                              const float* __restrict__ bq2,
                                              const float* __restrict__ bk2,
                                              float* __restrict__ beff) {
  int idx = blockIdx.x * 256 + threadIdx.x;
  if (idx >= 2048) return;
  int g = idx >> 9, a = idx & 511;
  const float* W = (g == 0) ? Wq1 : (g == 1) ? Wk1 : (g == 2) ? Wq2 : Wk2;
  const float* b = (g == 0) ? bq1 : (g == 1) ? bk1 : (g == 2) ? bq2 : bk2;
  beff[idx] = b[a] + W[(size_t)a * QKVD + 640];
}

// ---------------------------------------------------------------------------
// 3) Fused V weights:  wvh[m*8+h][c] = sum_d Wv_m[d*8+h, c] * w_m[d*8+h]
// ---------------------------------------------------------------------------
__global__ __launch_bounds__(256) void k_fuse_v(const float* __restrict__ Wv1,
                                                const float* __restrict__ Wv2,
                                                const float* __restrict__ wA,
                                                const float* __restrict__ wB,
                                                const float* __restrict__ bv2,
                                                float* __restrict__ wvh,
                                                float* __restrict__ bvw) {
  int idx = blockIdx.x * 256 + threadIdx.x;
  if (idx >= 2 * 8 * 641) return;
  int m = idx / (8 * 641);
  int r = idx % (8 * 641);
  int h = r / 641, c = r % 641;
  const float* Wv = m ? Wv2 : Wv1;
  const float* w  = m ? wB  : wA;
  float s = 0.f;
  if (c < 640) {
#pragma unroll 4
    for (int d = 0; d < 64; ++d) s += Wv[(size_t)(d * 8 + h) * QKVD + c] * w[d * 8 + h];
    wvh[(m * 8 + h) * KD + c] = s;
  } else {
#pragma unroll 4
    for (int d = 0; d < 64; ++d) {
      float t = Wv[(size_t)(d * 8 + h) * QKVD + 640];
      if (m) t += bv2[d * 8 + h];
      s += t * w[d * 8 + h];
    }
    bvw[m * 8 + h] = s;
  }
}

// ---------------------------------------------------------------------------
// 4) vw[m][b][h][k] = y[b,k,:] @ wvh[m*8+h] + bvw   (16 outputs per token)
// ---------------------------------------------------------------------------
__global__ __launch_bounds__(256) void k_vw(const _Float16* __restrict__ yh,
                                            const float* __restrict__ wvh,
                                            const float* __restrict__ bvw,
                                            float* __restrict__ vw) {
  __shared__ _Float16 ysh[16 * KD];                  // 20 KB
  int n0 = blockIdx.x * 16;
#pragma unroll
  for (int i = 0; i < 5; ++i) {
    int chunk = threadIdx.x + i * 256;               // 0..1279
    int row = chunk / 80;
    int kc  = (chunk % 80) * 8;
    cp_b128(ysh + row * KD + kc, yh + (size_t)(n0 + row) * KD + kc);
  }
  cp_b128_wait_all();
  __syncthreads();
  int pos = threadIdx.x >> 4;
  int mh  = threadIdx.x & 15;
  int nn  = n0 + pos;
  float s = bvw[mh];
  const float*    wp = wvh + mh * KD;
  const _Float16* yp = ysh + pos * KD;
#pragma unroll 8
  for (int c = 0; c < KD; ++c) s += (float)yp[c] * wp[c];
  int b = nn / HW, k = nn % HW;
  int m = mh >> 3, h = mh & 7;
  vw[(((size_t)m * BATCH + b) * 8 + h) * HW + k] = s;
}

// ---------------------------------------------------------------------------
// 5) GEMM: qk[50176][2048] = y(50176x640,f16) @ wcat^T + beff,  fp16 out.
//    128x128 block, 8 waves, each wave 32x64 (2x4 WMMA tiles), K step 32.
//    Double-buffered LDS; tiles staged by TDM (tensorcnt) or async copies
//    (asynccnt); threshold waits keep one tile pair in flight.
// ---------------------------------------------------------------------------
__global__ __launch_bounds__(256) void k_gemm(const _Float16* __restrict__ yh,
                                              const _Float16* __restrict__ wcat,
                                              const float* __restrict__ beff,
                                              _Float16* __restrict__ qk) {
  __shared__ _Float16 ysh[2][128 * LDY];             // 2 x 10 KB
  __shared__ _Float16 wsh[2][128 * LDY];             // 2 x 10 KB
  int col0 = blockIdx.x * 128;
  int row0 = blockIdx.y * 128;
  int lane = threadIdx.x & 31;
  int wave = threadIdx.x >> 5;
  int wm = wave >> 1;                                // 0..3 -> 32-row strip
  int wn = wave & 1;                                 // 0..1 -> 64-col strip

  auto stage = [&](int buf, int kk) {
    int k0 = kk * 32;
#if HAVE_TDM
    if (wave == 0) {                                 // one DMA pair per block
      tdm_load_tile(yh   + (size_t)row0 * KD + k0, ysh[buf]);
      tdm_load_tile(wcat + (size_t)col0 * KD + k0, wsh[buf]);
    }
#else
#pragma unroll
    for (int i = 0; i < 2; ++i) {
      int chunk = threadIdx.x + i * 256;             // 0..511
      int row = chunk >> 2;
      int kc  = (chunk & 3) << 3;
      cp_b128(&ysh[buf][row * LDY + kc], yh   + (size_t)(row0 + row) * KD + k0 + kc);
      cp_b128(&wsh[buf][row * LDY + kc], wcat + (size_t)(col0 + row) * KD + k0 + kc);
    }
#endif
  };

  v8f acc[2][4];
#pragma unroll
  for (int i = 0; i < 2; ++i)
#pragma unroll
    for (int j = 0; j < 4; ++j) acc[i][j] = v8f{};

  stage(0, 0);
  for (int kk = 0; kk < NKTILE; ++kk) {
    int cur = kk & 1;
    if (kk + 1 < NKTILE) {
      stage(cur ^ 1, kk + 1);                        // overlap next tile DMA
#if HAVE_TDM
      if (wave == 0) __builtin_amdgcn_s_wait_tensorcnt(2);   // tile kk landed
#elif HAVE_ASYNC
      __builtin_amdgcn_s_wait_asynccnt(4);                   // tile kk landed
#endif
    } else {
#if HAVE_TDM
      if (wave == 0) __builtin_amdgcn_s_wait_tensorcnt(0);
#elif HAVE_ASYNC
      __builtin_amdgcn_s_wait_asynccnt(0);
#endif
    }
    __syncthreads();                                 // cur buffer valid for all

    v16h a0 = ld_frag(ysh[cur], LDY, wm * 32 + 0,  127, 0, lane);
    v16h a1 = ld_frag(ysh[cur], LDY, wm * 32 + 16, 127, 0, lane);
#pragma unroll
    for (int j = 0; j < 4; ++j) {
      v16h bf = ld_frag(wsh[cur], LDY, wn * 64 + j * 16, 127, 0, lane);
      acc[0][j] = WMMA_F16(a0, bf, acc[0][j]);
      acc[1][j] = WMMA_F16(a1, bf, acc[1][j]);
    }
    __syncthreads();                                 // reads done before re-stage
  }

  int n  = lane & 15;
  int mo = (lane >> 4) << 3;
#pragma unroll
  for (int i = 0; i < 2; ++i) {
#pragma unroll
    for (int j = 0; j < 4; ++j) {
      int col = col0 + wn * 64 + j * 16 + n;
      float bias = beff[col];
      size_t rbase = (size_t)(row0 + wm * 32 + i * 16 + mo);
#pragma unroll
      for (int r = 0; r < 8; ++r)
        qk[(rbase + r) * 2048 + col] = (_Float16)(acc[i][j][r] + bias);
    }
  }
}

// ---------------------------------------------------------------------------
// 6) Attention: block = (mha, b, head, 16-query tile); 2 waves.
//    scores = Q K^T / scale via WMMA, softmax + vw-weighted sum per row.
// ---------------------------------------------------------------------------
__global__ __launch_bounds__(64) void k_attn(const _Float16* __restrict__ qk,
                                             const float* __restrict__ vw,
                                             const float* __restrict__ sc1,
                                             const float* __restrict__ sc2,
                                             float* __restrict__ headout) {
  __shared__ float sm[16 * SLD];                     // 13.4 KB
  int idx = blockIdx.x;
  int rt = idx % 13;  idx /= 13;
  int h  = idx % 8;   idx /= 8;
  int b  = idx % BATCH; idx /= BATCH;
  int m  = idx;                                      // 0 or 1
  int lane = threadIdx.x & 31;
  int wave = threadIdx.x >> 5;

  const _Float16* qbase = qk + ((size_t)b * HW) * 2048 + m * 1024 + h * 64;
  const _Float16* kbase = qbase + 512;
  float rs = 1.0f / (m == 0 ? sc1[0] : sc2[0]);
  int row0 = rt * 16;

  v16h a0 = ld_frag(qbase, 2048, row0, HW - 1, 0,  lane);
  v16h a1 = ld_frag(qbase, 2048, row0, HW - 1, 32, lane);
  int n  = lane & 15;
  int mo = (lane >> 4) << 3;

  for (int ct = wave; ct < 13; ct += 2) {
    v16h b0 = ld_frag(kbase, 2048, ct * 16, HW - 1, 0,  lane);
    v16h b1 = ld_frag(kbase, 2048, ct * 16, HW - 1, 32, lane);
    v8f c = v8f{};
    c = WMMA_F16(a0, b0, c);
    c = WMMA_F16(a1, b1, c);
#pragma unroll
    for (int r = 0; r < 8; ++r)
      sm[(mo + r) * SLD + ct * 16 + n] = c[r] * rs;
  }
  __syncthreads();

  int t = threadIdx.x;
  if (t < 16) {
    int p = row0 + t;
    if (p < HW) {
      const float* row = sm + t * SLD;
      float mx = -1e30f;
      for (int k = 0; k < HW; ++k) mx = fmaxf(mx, row[k]);
      const float* vwp = vw + (((size_t)m * BATCH + b) * 8 + h) * HW;
      float se = 0.f, ws = 0.f;
      for (int k = 0; k < HW; ++k) {
        float e = __expf(row[k] - mx);
        se += e;
        ws += e * vwp[k];
      }
      headout[((size_t)m * N_TOK + (size_t)b * HW + p) * 8 + h] = ws / se;
    }
  }
}

// ---------------------------------------------------------------------------
// 7) Final: A = exp(dotA); out = [y_tilde | A*x + dotB]; logdet = dotA*512.
// ---------------------------------------------------------------------------
__global__ __launch_bounds__(256) void k_final(const float* __restrict__ inputs,
                                               const float* __restrict__ headout,
                                               float* __restrict__ out,
                                               float* __restrict__ logdet) {
  int n = blockIdx.x;
  int t = threadIdx.x;
  const float* h1 = headout + (size_t)n * 8;
  const float* h2 = headout + ((size_t)N_TOK + n) * 8;
  float dA = 0.f, dB = 0.f;
#pragma unroll
  for (int h = 0; h < 8; ++h) { dA += h1[h]; dB += h2[h]; }
  float A = __expf(dA);
  int c = t * 4;
  float4 x = *(const float4*)(inputs + (size_t)n * CIN + c);
  float4 o;
  if (c < 512) {
    o = x;                                            // y_tilde passthrough
  } else {
    o.x = A * x.x + dB; o.y = A * x.y + dB;
    o.z = A * x.z + dB; o.w = A * x.w + dB;
  }
  *(float4*)(out + (size_t)n * CIN + c) = o;
  if (t == 0) logdet[n] = dA * 512.0f;
}

// ---------------------------------------------------------------------------
extern "C" void kernel_launch(void* const* d_in, const int* in_sizes, int n_in,
                              void* d_out, int out_size, void* d_ws, size_t ws_size,
                              hipStream_t stream) {
  (void)in_sizes; (void)n_in; (void)out_size; (void)ws_size;
  const float* inputs = (const float*)d_in[0];
  const float* cond   = (const float*)d_in[1];
  const float* Wq1 = (const float*)d_in[2];
  const float* bq1 = (const float*)d_in[3];
  const float* Wk1 = (const float*)d_in[4];
  const float* bk1 = (const float*)d_in[5];
  const float* Wv1 = (const float*)d_in[6];
  const float* Wq2 = (const float*)d_in[7];
  const float* bq2 = (const float*)d_in[8];
  const float* Wk2 = (const float*)d_in[9];
  const float* bk2 = (const float*)d_in[10];
  const float* Wv2 = (const float*)d_in[11];
  const float* bv2 = (const float*)d_in[12];
  const float* wA  = (const float*)d_in[13];
  const float* wB  = (const float*)d_in[14];
  const float* sc1 = (const float*)d_in[15];
  const float* sc2 = (const float*)d_in[16];

  char* ws = (char*)d_ws;
  size_t off = 0;
  auto carve = [&](size_t bytes) -> char* {
    char* p = ws + off;
    off = (off + bytes + 255) & ~(size_t)255;
    return p;
  };
  _Float16* yh      = (_Float16*)carve((size_t)N_TOK * KD * 2);      //  64.2 MB
  _Float16* wcat    = (_Float16*)carve((size_t)2048 * KD * 2);       //   2.6 MB
  float*    beff    = (float*)   carve((size_t)2048 * 4);
  float*    wvh     = (float*)   carve((size_t)16 * KD * 4);
  float*    bvw     = (float*)   carve((size_t)64);
  float*    vw      = (float*)   carve((size_t)2 * N_TOK * 8 * 4);   //   3.2 MB
  _Float16* qk      = (_Float16*)carve((size_t)N_TOK * 2048 * 2);    // 205.5 MB
  float*    headout = (float*)   carve((size_t)2 * N_TOK * 8 * 4);   //   3.2 MB

  float* out    = (float*)d_out;
  float* logdet = out + (size_t)N_TOK * CIN;

  k_build_y<<<15680, 256, 0, stream>>>(inputs, cond, yh);
  k_pack_w <<<640,   256, 0, stream>>>(Wq1, Wk1, Wq2, Wk2, wcat);
  k_beff   <<<8,     256, 0, stream>>>(Wq1, Wk1, Wq2, Wk2, bq1, bk1, bq2, bk2, beff);
  k_fuse_v <<<41,    256, 0, stream>>>(Wv1, Wv2, wA, wB, bv2, wvh, bvw);
  k_vw     <<<3136,  256, 0, stream>>>(yh, wvh, bvw, vw);
  k_gemm   <<<dim3(16, 392), 256, 0, stream>>>(yh, wcat, beff, qk);
  k_attn   <<<53248, 64,  0, stream>>>(qk, vw, sc1, sc2, headout);
  k_final  <<<N_TOK, 256, 0, stream>>>(inputs, headout, out, logdet);
}
